// GCNAE_74431783239742
// MI455X (gfx1250) — matrix operand order repaired and checked by reading, not compiled, not run on hardware
//
#include <hip/hip_runtime.h>

typedef __attribute__((ext_vector_type(16))) __bf16 v16bf;
typedef __attribute__((ext_vector_type(8)))  __bf16 v8bf;
typedef __attribute__((ext_vector_type(8)))  float  v8f;

constexpr int IN_F = 128;
constexpr int H_F  = 64;
constexpr int NN   = 10000;          // nodes (reference N)
constexpr int EE   = 320000;         // edges (reference E)
constexpr int NT   = NN / 16;        // 625 16-wide tiles per dim
constexpr int JREUSE = 8;            // J tiles per wave (A-fragment reuse)
constexpr int NJG  = (NT + JREUSE - 1) / JREUSE;   // 79
constexpr long long NTASKS = (long long)NT * NJG;  // 49375 wave-tasks

// ---------------- zero the aggregation buffer ----------------
__global__ void GCNAE_zero_f4(float4* __restrict__ p, int n4) {
    int i = blockIdx.x * blockDim.x + threadIdx.x;
    if (i < n4) p[i] = float4{0.f, 0.f, 0.f, 0.f};
}

// ---------------- edge gather + scatter-add (1 wave per edge) ----------------
// msgs = x[src] * w ; agg[dst] += msgs   (f32 atomics, agg is L2-resident)
__global__ void GCNAE_scatter(const float* __restrict__ x,
                              const int*   __restrict__ ei,
                              const float* __restrict__ ew,
                              float*       __restrict__ agg) {
    int wave = (int)((blockIdx.x * blockDim.x + threadIdx.x) >> 5);
    int lane = threadIdx.x & 31;
    if (wave >= EE) return;                      // wave-uniform guard
    int   src = ei[wave];
    int   dst = ei[EE + wave];
    float w   = ew[wave];
    float4 v  = ((const float4*)(x + (size_t)src * IN_F))[lane];
    float* ar = agg + (size_t)dst * IN_F + lane * 4;
    atomicAdd(ar + 0, v.x * w);
    atomicAdd(ar + 1, v.y * w);
    atomicAdd(ar + 2, v.z * w);
    atomicAdd(ar + 3, v.w * w);
}

// ---------------- z = agg @ W_rel + x @ W_root + b  (one row per block) ----------------
__global__ void GCNAE_zkern(const float* __restrict__ agg,
                            const float* __restrict__ x,
                            const float* __restrict__ Wrel,
                            const float* __restrict__ Wroot,
                            const float* __restrict__ b,
                            float*  __restrict__ z_f32,
                            __bf16* __restrict__ z_bf) {
    __shared__ float aggS[IN_F];
    __shared__ float xS[IN_F];
    int i = blockIdx.x;
    int t = threadIdx.x;                         // 0..63  (t == output feature h)
    aggS[t]      = agg[(size_t)i * IN_F + t];
    aggS[t + 64] = agg[(size_t)i * IN_F + t + 64];
    xS[t]        = x  [(size_t)i * IN_F + t];
    xS[t + 64]   = x  [(size_t)i * IN_F + t + 64];
    __syncthreads();
    float acc = b[t];
#pragma unroll 8
    for (int k = 0; k < IN_F; ++k) {
        acc = fmaf(aggS[k], Wrel [k * H_F + t], acc);
        acc = fmaf(xS[k],   Wroot[k * H_F + t], acc);
    }
    z_f32[(size_t)i * H_F + t] = acc;
    z_bf [(size_t)i * H_F + t] = (__bf16)acc;
}

// ---------------- adj = sigmoid(z @ z^T) via bf16 WMMA ----------------
// One wave computes a 16x16 tile (2x v_wmma_f32_16x16x32_bf16); A fragments
// reused across JREUSE consecutive J tiles. N is compile-time so the 8 C-row
// stores fold into base + immediate offsets.
__global__ void GCNAE_adj(const __bf16* __restrict__ z,
                          float*        __restrict__ adj) {
    long long task = ((long long)blockIdx.x * blockDim.x + threadIdx.x) >> 5;
    if (task >= NTASKS) return;                  // wave-uniform guard
    int lane = threadIdx.x & 31;
    int r    = lane & 15;
    int hi   = lane >> 4;                        // 0 or 1
    int ti   = (int)(task / NJG);
    int jg   = (int)(task % NJG);

    // A fragments: row ti*16 + r; contiguous 16-wide K-chunk per lane half (matches B).
    const __bf16* arow = z + (size_t)(ti * 16 + r) * H_F;
    v8bf  a0lo = *(const v8bf*)(arow + hi * 16);
    v8bf  a0hi = *(const v8bf*)(arow + hi * 16 + 8);
    v8bf  a1lo = *(const v8bf*)(arow + 32 + hi * 16);
    v8bf  a1hi = *(const v8bf*)(arow + 32 + hi * 16 + 8);
    v16bf a0 = __builtin_shufflevector(a0lo, a0hi, 0,1,2,3,4,5,6,7,8,9,10,11,12,13,14,15);
    v16bf a1 = __builtin_shufflevector(a1lo, a1hi, 0,1,2,3,4,5,6,7,8,9,10,11,12,13,14,15);

#pragma unroll
    for (int jj = 0; jj < JREUSE; ++jj) {
        int tj = jg * JREUSE + jj;
        if (tj >= NT) break;                     // wave-uniform
        // B fragments: column r of the tile == row tj*16+r of z; 16-wide K-chunk.
        const __bf16* brow = z + (size_t)(tj * 16 + r) * H_F;
        v16bf b0 = *(const v16bf*)(brow + hi * 16);
        v16bf b1 = *(const v16bf*)(brow + 32 + hi * 16);

        v8f c = {};
        c = __builtin_amdgcn_wmma_f32_16x16x32_bf16(false, a0, false, b0,
                                                    (short)0, c, false, false);
        c = __builtin_amdgcn_wmma_f32_16x16x32_bf16(false, a1, false, b1,
                                                    (short)0, c, false, false);

        // C layout: VGPR m, lanes 0-15 -> row m, lanes 16-31 -> row m+8; col = r.
        float* outBase = adj + (size_t)(ti * 16 + hi * 8) * NN + (size_t)tj * 16 + r;
#pragma unroll
        for (int m = 0; m < 8; ++m) {
            float v = c[m];
            float e = __builtin_amdgcn_exp2f(v * -1.44269504f);   // exp(-v)
            float s = __builtin_amdgcn_rcpf(1.0f + e);            // sigmoid
            __builtin_nontemporal_store(s, outBase + m * NN);     // imm offsets
        }
    }
}

extern "C" void kernel_launch(void* const* d_in, const int* in_sizes, int n_in,
                              void* d_out, int out_size, void* d_ws, size_t ws_size,
                              hipStream_t stream) {
    const float* x     = (const float*)d_in[0];
    const int*   ei    = (const int*)  d_in[1];
    const float* ew    = (const float*)d_in[2];
    const float* Wrel  = (const float*)d_in[3];
    const float* Wroot = (const float*)d_in[4];
    const float* b     = (const float*)d_in[5];

    float*  adj   = (float*)d_out;                           // [N, N]
    float*  z_f32 = (float*)d_out + (size_t)NN * NN;         // [N, H]
    float*  agg   = (float*)d_ws;                            // [N, IN] f32
    __bf16* z_bf  = (__bf16*)((char*)d_ws + (size_t)NN * IN_F * sizeof(float)); // [N, H] bf16

    // 1) agg = 0
    constexpr int n4 = NN * IN_F / 4;
    GCNAE_zero_f4<<<(n4 + 255) / 256, 256, 0, stream>>>((float4*)agg, n4);

    // 2) scatter-add weighted messages (8 edges per 256-thread block)
    GCNAE_scatter<<<(EE + 7) / 8, 256, 0, stream>>>(x, ei, ew, agg);

    // 3) node transform
    GCNAE_zkern<<<NN, 64, 0, stream>>>(agg, x, Wrel, Wroot, b, z_f32, z_bf);

    // 4) inner-product decoder with bf16 WMMA (8 waves / block)
    long long nBlocks = (NTASKS + 7) / 8;
    GCNAE_adj<<<(unsigned)nBlocks, 256, 0, stream>>>(z_bf, adj);
}
